// Attention_46780783788294
// MI455X (gfx1250) — matrix operand order repaired and checked by reading, not compiled, not run on hardware
//
#include <hip/hip_runtime.h>

// ---------------------------------------------------------------------------
// CDNA5 (gfx1250) attention block: QKV proj -> streaming-softmax attention
// (reference masks with 0.0, so ALL k positions contribute) -> output proj.
// All matmuls via v_wmma_f32_16x16x32_bf16 (wave32 WMMA, f32 accumulate).
// Round 3: B-tile staging is now fully coalesced (global_load_b128 row-major,
// transpose done on the LDS-store side); register double buffer kept.
// ---------------------------------------------------------------------------

typedef __attribute__((ext_vector_type(16))) __bf16 v16bf;
typedef __attribute__((ext_vector_type(8)))  __bf16 v8bf;
typedef __attribute__((ext_vector_type(8)))  float  v8f;
typedef __attribute__((ext_vector_type(4)))  float  v4f;

#define BATCH   2
#define S_LEN   2048
#define HEADS   16
#define DH      64
#define DM      1024
#define MROWS   (BATCH * S_LEN)   // 4096

__device__ __forceinline__ v16bf joinbf(v8bf lo, v8bf hi) {
    v16bf r;
#pragma unroll
    for (int i = 0; i < 8; ++i) { r[i] = lo[i]; r[i + 8] = hi[i]; }
    return r;
}
__device__ __forceinline__ v8f wmma_bf16(v16bf a, v16bf b, v8f c) {
    return __builtin_amdgcn_wmma_f32_16x16x32_bf16(false, a, false, b,
                                                   (short)0, c, false, false);
}

// ---------------------------------------------------------------------------
// Generic WMMA GEMM:  C[M,N] = A[M,K] * B(k,n) + bias[n]
//   bmode 0: B is W_{Q,K,V} layout [H, K, 64]  (BN=64 == one head, so each
//            k-row of the tile is 64 contiguous floats)
//   bmode 1: B is row-major [K, N]
//   smode 0: store bf16 [B,H,S,64]     (Q, K for attention)
//   smode 1: store bf16 [B,H,64,S]     (V transposed for attention PV WMMA)
//   smode 2: store f32 row-major [M,N] (final output)
// Block tile 128x64, BK=32, 256 threads (8 waves), wave tile 32x32.
// k-loop is register-double-buffered; all global traffic is b128-coalesced,
// the B transpose happens on the LDS-store side.
// ---------------------------------------------------------------------------
#define BM 128
#define BN 64
#define BK 32

__global__ __launch_bounds__(256) void gemm_wmma_kernel(
    const float* __restrict__ A, const float* __restrict__ Bsrc,
    const float* __restrict__ bias, void* __restrict__ Cout,
    int M, int N, int K, int bmode, int smode)
{
    __shared__ __attribute__((aligned(32))) __bf16 As[BM * BK];   // [row][k]
    __shared__ __attribute__((aligned(32))) __bf16 Bs[BN * BK];   // [n][k]

    const int tid  = threadIdx.x;
    const int lane = tid & 31;
    const int wave = tid >> 5;
    const int half = lane >> 4;
    const int ln   = lane & 15;
    const int wm   = wave >> 1;       // 0..3
    const int wn   = wave & 1;        // 0..1
    const int m0   = blockIdx.y * BM;
    const int n0   = blockIdx.x * BN;

    // A stage: thread -> (row, 16-float slab)
    const int arow = tid >> 1, acol = (tid & 1) << 4;
    const float* Abase = A + (size_t)(m0 + arow) * K + acol;

    // B stage: thread -> (k-row, 8-float slab); fully contiguous in memory
    const int bkrow = tid >> 3;            // 0..31
    const int bcol8 = (tid & 7) << 3;      // 0..56
    const size_t bstride = (bmode == 0) ? (size_t)64 : (size_t)N; // elems/k-row
    const float* Bbase = Bsrc + ((bmode == 0)
        ? (size_t)(n0 >> 6) * ((size_t)K * 64) + (size_t)bkrow * 64 + bcol8
        : (size_t)bkrow * N + n0 + bcol8);

    v8f acc[2][2] = {};

    // ---- prologue: prefetch tile 0 into registers
    v4f ra[4], rb[2];
#pragma unroll
    for (int i = 0; i < 4; ++i)
        ra[i] = *reinterpret_cast<const v4f*>(Abase + 4 * i);
#pragma unroll
    for (int i = 0; i < 2; ++i)
        rb[i] = *reinterpret_cast<const v4f*>(Bbase + 4 * i);

    for (int kb = 0; kb < K; kb += BK) {
        // ---- commit staged registers to LDS (hardware f32->bf16 cvt)
#pragma unroll
        for (int i = 0; i < 16; ++i)
            As[arow * BK + acol + i] = (__bf16)ra[i >> 2][i & 3];
#pragma unroll
        for (int i = 0; i < 8; ++i)   // transpose on the LDS side: Bs[n][k]
            Bs[(bcol8 + i) * BK + bkrow] = (__bf16)rb[i >> 2][i & 3];
        __syncthreads();

        // ---- issue next tile's global loads (overlap with WMMA below)
        if (kb + BK < K) {
            const float* Ag = Abase + kb + BK;
#pragma unroll
            for (int i = 0; i < 4; ++i)
                ra[i] = *reinterpret_cast<const v4f*>(Ag + 4 * i);
            const float* Bg = Bbase + (size_t)(kb + BK) * bstride;
#pragma unroll
            for (int i = 0; i < 2; ++i)
                rb[i] = *reinterpret_cast<const v4f*>(Bg + 4 * i);
        }

        // ---- fragments (16x16x32 bf16; contiguous k = 16*half + s in A and B alike)
        v16bf af[2], bfr[2];
#pragma unroll
        for (int mt = 0; mt < 2; ++mt) {
            const int row = wm * 32 + mt * 16 + ln;
            v8bf lo = *reinterpret_cast<const v8bf*>(&As[row * BK + half * 16]);
            v8bf hi = *reinterpret_cast<const v8bf*>(&As[row * BK + half * 16 + 8]);
            af[mt] = joinbf(lo, hi);
        }
#pragma unroll
        for (int nt = 0; nt < 2; ++nt) {
            const int col = wn * 32 + nt * 16 + ln;
            bfr[nt] = *reinterpret_cast<const v16bf*>(&Bs[col * BK + half * 16]);
        }
#pragma unroll
        for (int mt = 0; mt < 2; ++mt)
#pragma unroll
            for (int nt = 0; nt < 2; ++nt)
                acc[mt][nt] = wmma_bf16(af[mt], bfr[nt], acc[mt][nt]);
        __syncthreads();
    }

    // ---- epilogue: bias + store in requested layout
#pragma unroll
    for (int mt = 0; mt < 2; ++mt) {
#pragma unroll
        for (int nt = 0; nt < 2; ++nt) {
            const int gcol = n0 + wn * 32 + nt * 16 + ln;
            const float bv = bias[gcol];
#pragma unroll
            for (int r = 0; r < 8; ++r) {
                const int grow = m0 + wm * 32 + mt * 16 + r + half * 8;
                const float v  = acc[mt][nt][r] + bv;
                if (smode == 2) {
                    ((float*)Cout)[(size_t)grow * N + gcol] = v;
                } else {
                    const int bb = grow >> 11, ss = grow & 2047;   // S = 2048
                    const int hh = gcol >> 6,  dd = gcol & 63;     // D = 64
                    size_t idx = (smode == 0)
                        ? (((size_t)bb * HEADS + hh) * S_LEN + ss) * DH + dd
                        : (((size_t)bb * HEADS + hh) * DH + dd) * S_LEN + ss;
                    ((__bf16*)Cout)[idx] = (__bf16)v;
                }
            }
        }
    }
}

// ---------------------------------------------------------------------------
// Attention: one wave per (b, h, 16-query tile). k-step = 32.
// scores = Q K^T / 8; masked (k>q) entries forced to 0.0 (reference semantics),
// online softmax over ALL 2048 columns, z += P * V via WMMA with P staged
// through LDS to re-layout C-frag -> A-frag.
// ---------------------------------------------------------------------------
__global__ __launch_bounds__(32) void attention_kernel(
    const __bf16* __restrict__ Qb,   // bf16 [B,H,S,64]
    const __bf16* __restrict__ Kb,   // bf16 [B,H,S,64]
    const __bf16* __restrict__ Vt,   // bf16 [B,H,64,S]
    float* __restrict__ Z)           // f32  [B,S,H,64]
{
    __shared__ __attribute__((aligned(32))) __bf16 Pbuf[16 * 32];

    const int lane = threadIdx.x & 31;
    const int half = lane >> 4;
    const int ln   = lane & 15;
    const int q0   = blockIdx.x * 16;
    const int h    = blockIdx.y;
    const int b    = blockIdx.z;
    const int bh   = b * HEADS + h;

    // Q A-fragments (d = 0..31 and 32..63), one per d-block of 32
    const __bf16* Qrow = Qb + ((size_t)bh * S_LEN + q0 + ln) * DH;
    v16bf aq0 = joinbf(*reinterpret_cast<const v8bf*>(Qrow + half * 16),
                       *reinterpret_cast<const v8bf*>(Qrow + half * 16 + 8));
    v16bf aq1 = joinbf(*reinterpret_cast<const v8bf*>(Qrow + 32 + half * 16),
                       *reinterpret_cast<const v8bf*>(Qrow + 32 + half * 16 + 8));

    float mrow[8], lrow[8];
#pragma unroll
    for (int r = 0; r < 8; ++r) { mrow[r] = -1e30f; lrow[r] = 0.0f; }
    v8f zacc[4] = {};

    const __bf16* Kbase  = Kb + (size_t)bh * S_LEN * DH;
    const __bf16* Vtbase = Vt + (size_t)bh * DH * S_LEN;

    for (int k0 = 0; k0 < S_LEN; k0 += 32) {
        // ---- scores for 16q x 32k: two 16x16 tiles, accumulate over d
        v8f s0 = {}, s1 = {};
#pragma unroll
        for (int t = 0; t < 2; ++t) {
            const __bf16* Krow = Kbase + (size_t)(k0 + t * 16 + ln) * DH;
            v16bf kf0 = *reinterpret_cast<const v16bf*>(Krow + half * 16);
            v16bf kf1 = *reinterpret_cast<const v16bf*>(Krow + 32 + half * 16);
            if (t == 0) { s0 = wmma_bf16(aq0, kf0, s0); s0 = wmma_bf16(aq1, kf1, s0); }
            else        { s1 = wmma_bf16(aq0, kf0, s1); s1 = wmma_bf16(aq1, kf1, s1); }
        }

        // ---- scale + mask-to-zero (reference uses 0.0, not -inf)
        float rm[8];
#pragma unroll
        for (int r = 0; r < 8; ++r) {
            const int qg = q0 + r + half * 8;
            s0[r] = ((k0 + ln)      <= qg) ? s0[r] * 0.125f : 0.0f;
            s1[r] = ((k0 + 16 + ln) <= qg) ? s1[r] * 0.125f : 0.0f;
            rm[r] = fmaxf(s0[r], s1[r]);
        }
        // row max across the 16 lanes of each half
#pragma unroll
        for (int off = 1; off < 16; off <<= 1)
#pragma unroll
            for (int r = 0; r < 8; ++r)
                rm[r] = fmaxf(rm[r], __shfl_xor(rm[r], off, 32));

        // ---- online softmax update + stage P (bf16) to LDS in C layout
#pragma unroll
        for (int r = 0; r < 8; ++r) {
            const float mnew = fmaxf(mrow[r], rm[r]);
            const float corr = __expf(mrow[r] - mnew);
            mrow[r] = mnew;
            const float p0 = __expf(s0[r] - mnew);
            const float p1 = __expf(s1[r] - mnew);
            lrow[r] = lrow[r] * corr + p0 + p1;   // per-lane partial row sum
#pragma unroll
            for (int nt = 0; nt < 4; ++nt) zacc[nt][r] *= corr;
            Pbuf[(r + half * 8) * 32 + ln]      = (__bf16)p0;
            Pbuf[(r + half * 8) * 32 + 16 + ln] = (__bf16)p1;
        }
        asm volatile("s_wait_dscnt 0" ::: "memory");   // single-wave LDS RAW

        // ---- P as A-fragment (16x32), V^T columns as B-fragments
        v16bf pf = joinbf(*reinterpret_cast<const v8bf*>(&Pbuf[ln * 32 + half * 16]),
                          *reinterpret_cast<const v8bf*>(&Pbuf[ln * 32 + half * 16 + 8]));
#pragma unroll
        for (int nt = 0; nt < 4; ++nt) {
            v16bf vf = *reinterpret_cast<const v16bf*>(
                Vtbase + (size_t)(nt * 16 + ln) * S_LEN + k0 + half * 16);
            zacc[nt] = wmma_bf16(pf, vf, zacc[nt]);
        }
    }

    // ---- finalize: reduce l across lanes, normalize, store z [B,S,H,64] f32
#pragma unroll
    for (int off = 1; off < 16; off <<= 1)
#pragma unroll
        for (int r = 0; r < 8; ++r)
            lrow[r] += __shfl_xor(lrow[r], off, 32);

#pragma unroll
    for (int r = 0; r < 8; ++r) {
        const float inv = 1.0f / lrow[r];
        float* Zrow = Z + (((size_t)b * S_LEN + q0 + r + half * 8) * HEADS + h) * DH;
#pragma unroll
        for (int nt = 0; nt < 4; ++nt)
            Zrow[nt * 16 + ln] = zacc[nt][r] * inv;
    }
}

// ---------------------------------------------------------------------------
extern "C" void kernel_launch(void* const* d_in, const int* in_sizes, int n_in,
                              void* d_out, int out_size, void* d_ws, size_t ws_size,
                              hipStream_t stream) {
    (void)in_sizes; (void)n_in; (void)out_size; (void)ws_size;

    const float* x   = (const float*)d_in[0];
    const float* W_Q = (const float*)d_in[1];
    const float* b_Q = (const float*)d_in[2];
    const float* W_K = (const float*)d_in[3];
    const float* b_K = (const float*)d_in[4];
    const float* W_V = (const float*)d_in[5];
    const float* b_V = (const float*)d_in[6];
    const float* W_O = (const float*)d_in[7];
    const float* b_O = (const float*)d_in[8];
    float* out = (float*)d_out;

    // workspace carve-up: Q,K,Vt bf16 (8 MB each) + z f32 (16 MB) = 40 MB
    __bf16* Qb = (__bf16*)d_ws;
    __bf16* Kb = Qb + (size_t)MROWS * DM;
    __bf16* Vt = Kb + (size_t)MROWS * DM;
    float*  Zf = (float*)(Vt + (size_t)MROWS * DM);

    dim3 gG(DM / BN, MROWS / BM);           // (16, 32)
    dim3 bG(256);
    // QKV projections (bmode 0 = [H,K,64] weight layout)
    gemm_wmma_kernel<<<gG, bG, 0, stream>>>(x, W_Q, b_Q, Qb, MROWS, DM, DM, 0, 0);
    gemm_wmma_kernel<<<gG, bG, 0, stream>>>(x, W_K, b_K, Kb, MROWS, DM, DM, 0, 0);
    gemm_wmma_kernel<<<gG, bG, 0, stream>>>(x, W_V, b_V, Vt, MROWS, DM, DM, 0, 1);

    dim3 gA(S_LEN / 16, HEADS, BATCH);      // (128, 16, 2)
    attention_kernel<<<gA, dim3(32), 0, stream>>>(Qb, Kb, Vt, Zf);

    // output projection (bmode 1 = row-major [K,N] weight, smode 2 = f32 out)
    gemm_wmma_kernel<<<gG, bG, 0, stream>>>(Zf, W_O, b_O, out, MROWS, DM, DM, 1, 2);
}
